// WaveRNN_10110353014968
// MI455X (gfx1250) — matrix operand order repaired and checked by reading, not compiled, not run on hardware
//
#include <hip/hip_runtime.h>
#include <math.h>

typedef __bf16 bf16;
typedef __attribute__((ext_vector_type(16))) __bf16 v16bf;
typedef __attribute__((ext_vector_type(8)))  __bf16 v8bf;
typedef __attribute__((ext_vector_type(8)))  float  v8f;

#define WMMA_BF16(a,b,c) __builtin_amdgcn_wmma_f32_16x16x32_bf16(false,(a),false,(b),(short)0,(c),false,false)

// ---------------- constants ----------------
#define BATCH   4
#define FEAT    80
#define FRAMES  54
#define LCONV   50
#define CCH     128
#define HDIM    512
#define TLEN    12800
#define BT      (BATCH*TLEN)
#define AUXD    32
#define GRU_NWG     8
#define GRU_THREADS 384
// per-WG packed whh: 12 tiles x 16 kIters x 32 lanes x 16 bf16
#define WHH_WG_ELEMS (12*16*32*16)
#define WHH_WG_BYTES (WHH_WG_ELEMS*2)          // 196608
#define GRU_LDS_BYTES (WHH_WG_BYTES + 16*HDIM*2 + 3*64*BATCH*4)   // 216064

// ---------------- small fp32 kernels (mel resnet / upsample / packing) ----------------

__global__ void k_conv_in(const float* __restrict__ mels, const float* __restrict__ w,
                          const float* __restrict__ g, const float* __restrict__ bb,
                          float* __restrict__ out)
{
    int idx = blockIdx.x * blockDim.x + threadIdx.x;
    if (idx >= BATCH*CCH*LCONV) return;
    int l = idx % LCONV; int o = (idx / LCONV) % CCH; int b = idx / (LCONV*CCH);
    float s = 0.f;
    for (int i = 0; i < FEAT; i++) {
        const float* mrow = mels + (b*FEAT + i)*FRAMES + l;
        const float* wrow = w + (o*FEAT + i)*5;
        #pragma unroll
        for (int k = 0; k < 5; k++) s += mrow[k] * wrow[k];
    }
    float inv = g[o] * rsqrtf(1.f + 1e-5f);
    float v = s * inv + bb[o];
    out[idx] = v > 0.f ? v : 0.f;
}

__global__ void k_mm128(const float* __restrict__ in, const float* __restrict__ W,
                        const float* __restrict__ g, const float* __restrict__ bb,
                        const float* __restrict__ res, float* __restrict__ out, int relu)
{
    int idx = blockIdx.x * blockDim.x + threadIdx.x;
    if (idx >= BATCH*CCH*LCONV) return;
    int l = idx % LCONV; int o = (idx / LCONV) % CCH; int b = idx / (LCONV*CCH);
    float s = 0.f;
    const float* irow = in + (b*CCH)*LCONV + l;
    const float* wrow = W + o*CCH;
    for (int c = 0; c < CCH; c++) s += irow[c*LCONV] * wrow[c];
    float inv = g ? g[o] * rsqrtf(1.f + 1e-5f) : 1.f;
    float v = s * inv + bb[o];
    if (relu) v = v > 0.f ? v : 0.f;
    if (res) v += res[idx];
    out[idx] = v;
}

__global__ void k_upsample(const float* __restrict__ in, float* __restrict__ out,
                           const float* __restrict__ w, int Lin, int Lout,
                           int s, int kw, int crop, int outT)
{
    int idx = blockIdx.x * blockDim.x + threadIdx.x;
    if (idx >= BATCH*FEAT*Lout) return;
    int t = idx % Lout; int f = (idx / Lout) % FEAT; int b = idx / (Lout*FEAT);
    const float* irow = in + (b*FEAT + f)*Lin;
    int te = t + crop;
    float acc = 0.f;
    for (int k = 0; k < kw; k++) {
        int r = te + k - s;
        if (r >= 0 && r < Lin*s) acc += w[k] * irow[r / s];
    }
    if (outT) out[((size_t)b*TLEN + t)*FEAT + f] = acc;
    else      out[(size_t)(b*FEAT + f)*Lout + t] = acc;
}

__global__ void k_cvt_w(const float* __restrict__ in, bf16* __restrict__ out, int N, int K, int Kp)
{
    int idx = blockIdx.x * blockDim.x + threadIdx.x;
    if (idx >= N*Kp) return;
    int k = idx % Kp; int n = idx / Kp;
    out[idx] = (k < K) ? (bf16)in[(size_t)n*K + k] : (bf16)0.f;
}

// pack whh [1536][512] f32 -> per-WG WMMA B-fragment order:
// out[(((wg*12 + v)*16 + kIt)*32 + lane)*16 + i] = whh[g][k]
//   g = (v/4)*512 + wg*64 + (v%4)*16 + (lane&15) ; k = kIt*32 + 16*(lane>>4) + i
__global__ void k_pack_whh(const float* __restrict__ whh, bf16* __restrict__ out)
{
    int idx = blockIdx.x * blockDim.x + threadIdx.x;
    if (idx >= GRU_NWG*WHH_WG_ELEMS) return;
    int i    = idx & 15;
    int lane = (idx >> 4) & 31;
    int kIt  = (idx >> 9) & 15;
    int v    = (idx >> 13) % 12;
    int wg   = idx / (12*16*32*16);
    int g = (v >> 2)*HDIM + wg*64 + (v & 3)*16 + (lane & 15);
    int k = kIt*32 + 16*(lane >> 4) + i;
    out[idx] = (bf16)whh[(size_t)g*HDIM + k];
}

__global__ void k_ain1(const float* __restrict__ x, const float* __restrict__ mup,
                       const float* __restrict__ aux, bf16* __restrict__ out)
{
    size_t idx = (size_t)blockIdx.x * blockDim.x + threadIdx.x;
    if (idx >= (size_t)BT*128) return;
    int c = (int)(idx & 127); size_t bt = idx >> 7;
    int t = (int)(bt % TLEN); int b = (int)(bt / TLEN);
    float v;
    if (c == 0)       v = x[bt];
    else if (c < 81)  v = mup[bt*FEAT + (c-1)];
    else if (c < 113) v = aux[((size_t)b*CCH + (c-81))*LCONV + (t >> 8)];
    else              v = 0.f;
    out[idx] = (bf16)v;
}

__global__ void k_packaux(const float* __restrict__ aux, bf16* __restrict__ dst, int ld, int seg)
{
    size_t idx = (size_t)blockIdx.x * blockDim.x + threadIdx.x;
    if (idx >= (size_t)BT*AUXD) return;
    int c = (int)(idx % AUXD); size_t bt = idx / AUXD;
    int t = (int)(bt % TLEN); int b = (int)(bt / TLEN);
    dst[bt*ld + HDIM + c] = (bf16)aux[((size_t)b*CCH + seg*AUXD + c)*LCONV + (t >> 8)];
}

// ---------------- WMMA fragment helpers ----------------

__device__ __forceinline__ v16bf load_afrag(const bf16* __restrict__ A, int lda, int m0, int k0, int lane)
{
    int row = m0 + (lane & 15);
    int half = lane >> 4;
    const bf16* p = A + (size_t)row*lda + k0 + 16*half;
    v8bf lo = *(const v8bf*)p;
    v8bf hi = *(const v8bf*)(p + 8);
    v16bf a;
    #pragma unroll
    for (int i = 0; i < 8; i++) { a[i] = lo[i]; a[i+8] = hi[i]; }
    return a;
}

__device__ __forceinline__ v16bf load_bfrag(const bf16* __restrict__ W, int ldw, int n0, int k0, int lane)
{
    const bf16* p = W + (size_t)(n0 + (lane & 15))*ldw + k0 + 16*(lane >> 4);
    return *(const v16bf*)p;
}

// ---------------- bf16 WMMA GEMM: C[M,N] = A[M,Kp] * W[N,Kp]^T + bias ----------------
// each wave computes a 32x64 tile: 2 A-frags x 4 B-frags, 8 accumulators.
__global__ __launch_bounds__(256) void k_gemm(const bf16* __restrict__ A, const bf16* __restrict__ W,
                                              const float* __restrict__ bias,
                                              float* __restrict__ outF, bf16* __restrict__ outB,
                                              int M, int N, int Kp, int ldc, int relu)
{
    int lane = threadIdx.x & 31;
    int wid = blockIdx.x * (blockDim.x >> 5) + (threadIdx.x >> 5);
    int tilesN = N >> 6;
    int m0 = (wid / tilesN) << 5;
    int n0 = (wid % tilesN) << 6;
    if (m0 >= M) return;

    v8f acc[2][4] = {{v8f{}, v8f{}, v8f{}, v8f{}}, {v8f{}, v8f{}, v8f{}, v8f{}}};
    for (int k0 = 0; k0 < Kp; k0 += 32) {
        v16bf a0 = load_afrag(A, Kp, m0,      k0, lane);
        v16bf a1 = load_afrag(A, Kp, m0 + 16, k0, lane);
        #pragma unroll
        for (int j = 0; j < 4; j++) {
            v16bf b = load_bfrag(W, Kp, n0 + 16*j, k0, lane);
            acc[0][j] = WMMA_BF16(a0, b, acc[0][j]);
            acc[1][j] = WMMA_BF16(a1, b, acc[1][j]);
        }
    }
    int rowHalf = (lane >> 4) * 8;
    #pragma unroll
    for (int mi = 0; mi < 2; mi++) {
        #pragma unroll
        for (int j = 0; j < 4; j++) {
            int n = n0 + 16*j + (lane & 15);
            float bv = bias ? bias[n] : 0.f;
            #pragma unroll
            for (int r = 0; r < 8; r++) {
                int m = m0 + 16*mi + rowHalf + r;
                float v = acc[mi][j][r] + bv;
                if (relu) v = v > 0.f ? v : 0.f;
                if (outF) outF[(size_t)m*ldc + n] = v;
                if (outB) outB[(size_t)m*ldc + n] = (bf16)v;
            }
        }
    }
}

// ---------------- persistent GRU scan ----------------
// 8 WGs; WG wg owns h columns [wg*64, wg*64+64). whh slice (192 rows) lives in LDS,
// preloaded once via global_load_async_to_lds_b128. Per step only the 4KB h broadcast
// moves through global memory (also async-to-LDS). Global barrier per step.
__global__ __launch_bounds__(GRU_THREADS) void k_gru(
    const float* __restrict__ xp,            // [b*T+t][1536] f32 (x@wih + bih)
    const bf16*  __restrict__ whhP,          // per-WG packed fragments
    const float* __restrict__ bhh,
    const bf16*  __restrict__ resB, int ldres,
    bf16* __restrict__ out, int ldout,
    float* __restrict__ hG,
    bf16* __restrict__ hGb0, bf16* __restrict__ hGb1,
    unsigned* __restrict__ cnt, int T)
{
    extern __shared__ char smem[];
    bf16*  whhL = (bf16*)smem;                                   // 196608 B
    bf16*  hL   = (bf16*)(smem + WHH_WG_BYTES);                  // 16384 B
    float* ghL  = (float*)(smem + WHH_WG_BYTES + 16*HDIM*2);     // 3072 B
    unsigned ldsBase = (unsigned)(uintptr_t)smem;

    int tid = threadIdx.x, lane = tid & 31, wave = tid >> 5, wg = blockIdx.x;
    int gate = wave >> 2, tc = wave & 3;

    // zero h tile (rows 4..15 stay zero forever)
    for (int i = tid; i < 16*HDIM; i += GRU_THREADS) hL[i] = (bf16)0.f;

    // async preload of this WG's whh slice into LDS (12288 x 16B chunks)
    {
        const char* wsrc = (const char*)(whhP + (size_t)wg*WHH_WG_ELEMS);
        for (int c = tid; c < WHH_WG_BYTES/16; c += GRU_THREADS) {
            unsigned ldsOff = ldsBase + (unsigned)c*16u;
            unsigned long long ga = (unsigned long long)(uintptr_t)(wsrc) + (unsigned long long)c*16ull;
            asm volatile("global_load_async_to_lds_b128 %0, %1, off"
                         :: "v"(ldsOff), "v"(ga) : "memory");
        }
    }
    asm volatile("s_wait_asynccnt 0" ::: "memory");
    __syncthreads();

    for (int t = 0; t < T; t++) {
        const bf16* hin  = (t & 1) ? hGb1 : hGb0;
        bf16*       hout = (t & 1) ? hGb0 : hGb1;
        // async broadcast-load h (4 rows x 512 bf16 = 4096 B) into hL rows 0..3
        if (tid < 256) {
            unsigned ldsOff = ldsBase + (unsigned)WHH_WG_BYTES + (unsigned)tid*16u;
            unsigned long long ga = (unsigned long long)(uintptr_t)hin + (unsigned long long)tid*16ull;
            asm volatile("global_load_async_to_lds_b128 %0, %1, off"
                         :: "v"(ldsOff), "v"(ga) : "memory");
        }
        asm volatile("s_wait_asynccnt 0" ::: "memory");
        __syncthreads();

        // gh partial: (16x512)x(512x16) via 16 WMMA k-steps, B from LDS
        v8f acc = v8f{};
        int row = lane & 15, half = lane >> 4;
        #pragma unroll
        for (int kIt = 0; kIt < 16; kIt++) {
            int k0 = kIt*32;
            const bf16* p = &hL[row*HDIM + k0 + 16*half];
            v8bf lo = *(const v8bf*)p;
            v8bf hi = *(const v8bf*)(p + 8);
            v16bf a;
            #pragma unroll
            for (int i = 0; i < 8; i++) { a[i] = lo[i]; a[i+8] = hi[i]; }
            v16bf b = *(const v16bf*)&whhL[((wave*16 + kIt)*32 + lane)*16];
            acc = WMMA_BF16(a, b, acc);
        }
        if (lane < 16) {
            #pragma unroll
            for (int r = 0; r < BATCH; r++)
                ghL[(gate*64 + tc*16 + lane)*BATCH + r] = acc[r];
        }
        __syncthreads();

        if (tid < 64*BATCH) {
            int b = tid & (BATCH-1), c = tid / BATCH;
            int j = wg*64 + c;
            size_t rowi = (size_t)b*T + t;
            const float* xrow = xp + rowi*1536;
            float ghr = ghL[(c      )*BATCH + b] + bhh[j];
            float ghz = ghL[(64  + c)*BATCH + b] + bhh[HDIM + j];
            float ghn = ghL[(128 + c)*BATCH + b] + bhh[2*HDIM + j];
            float r = 1.f / (1.f + __expf(-(xrow[j] + ghr)));
            float z = 1.f / (1.f + __expf(-(xrow[HDIM + j] + ghz)));
            float n = tanhf(xrow[2*HDIM + j] + r*ghn);
            float hold = hG[b*HDIM + j];
            float hnew = (1.f - z)*n + z*hold;
            hG[b*HDIM + j] = hnew;
            hout[b*HDIM + j] = (bf16)hnew;
            out[rowi*ldout + j] = (bf16)(hnew + (float)resB[rowi*ldres + j]);
        }
        __syncthreads();
        if (tid == 0) {
            __threadfence();
            atomicAdd(cnt, 1u);
            unsigned target = (unsigned)GRU_NWG * (unsigned)(t + 1);
            volatile unsigned* vc = cnt;
            while (*vc < target) { }
            __threadfence();
        }
        __syncthreads();
    }
}

// ---------------- host side ----------------

static inline int blocks_for(size_t n, int bs) { return (int)((n + bs - 1) / bs); }

extern "C" void kernel_launch(void* const* d_in, const int* in_sizes, int n_in,
                              void* d_out, int out_size, void* d_ws, size_t ws_size,
                              hipStream_t stream)
{
    const float* x        = (const float*)d_in[0];
    const float* mels     = (const float*)d_in[1];
    const float* conv_in_w= (const float*)d_in[2];
    const float* bn0_g    = (const float*)d_in[3];
    const float* bn0_b    = (const float*)d_in[4];
    const float* res_c1   = (const float*)d_in[5];
    const float* res_c2   = (const float*)d_in[6];
    const float* rbn1g    = (const float*)d_in[7];
    const float* rbn1b    = (const float*)d_in[8];
    const float* rbn2g    = (const float*)d_in[9];
    const float* rbn2b    = (const float*)d_in[10];
    const float* conv_out_w = (const float*)d_in[11];
    const float* conv_out_b = (const float*)d_in[12];
    const float* up_w0    = (const float*)d_in[13];
    const float* up_w1    = (const float*)d_in[14];
    const float* up_w2    = (const float*)d_in[15];
    const float* I_w      = (const float*)d_in[16];
    const float* I_b      = (const float*)d_in[17];
    const float* r1_wih   = (const float*)d_in[18];
    const float* r1_whh   = (const float*)d_in[19];
    const float* r1_bih   = (const float*)d_in[20];
    const float* r1_bhh   = (const float*)d_in[21];
    const float* r2_wih   = (const float*)d_in[22];
    const float* r2_whh   = (const float*)d_in[23];
    const float* r2_bih   = (const float*)d_in[24];
    const float* r2_bhh   = (const float*)d_in[25];
    const float* fc1_w    = (const float*)d_in[26];
    const float* fc1_b    = (const float*)d_in[27];
    const float* fc2_w    = (const float*)d_in[28];
    const float* fc2_b    = (const float*)d_in[29];
    const float* fc3_w    = (const float*)d_in[30];
    const float* fc3_b    = (const float*)d_in[31];

    char* ws = (char*)d_ws;
    size_t off = 0;
    auto alloc = [&](size_t bytes) -> char* {
        char* p = ws + off;
        off = (off + bytes + 255) & ~(size_t)255;
        return p;
    };

    float* bufA  = (float*)alloc(BATCH*CCH*LCONV*4);
    float* bufB  = (float*)alloc(BATCH*CCH*LCONV*4);
    float* bufT  = (float*)alloc(BATCH*CCH*LCONV*4);
    float* aux_s = (float*)alloc(BATCH*CCH*LCONV*4);
    float* u1    = (float*)alloc((size_t)BATCH*FEAT*216*4);
    float* u2    = (float*)alloc((size_t)BATCH*FEAT*1728*4);
    float* m_up  = (float*)alloc((size_t)BT*FEAT*4);

    bf16* Iw_b    = (bf16*)alloc((size_t)512*128*2);
    bf16* r1wih_b = (bf16*)alloc((size_t)1536*512*2);
    bf16* r2wih_b = (bf16*)alloc((size_t)1536*544*2);
    bf16* fc1w_b  = (bf16*)alloc((size_t)512*544*2);
    bf16* fc2w_b  = (bf16*)alloc((size_t)512*544*2);
    bf16* fc3w_b  = (bf16*)alloc((size_t)1024*512*2);
    bf16* whh1P   = (bf16*)alloc((size_t)GRU_NWG*WHH_WG_ELEMS*2);
    bf16* whh2P   = (bf16*)alloc((size_t)GRU_NWG*WHH_WG_ELEMS*2);

    bf16* Ain1 = (bf16*)alloc((size_t)BT*128*2);
    bf16* h0   = (bf16*)alloc((size_t)BT*512*2);
    float* xp  = (float*)alloc((size_t)BT*1536*4);      // reused for both GRUs
    bf16* Ain2 = (bf16*)alloc((size_t)BT*544*2);
    bf16* Ain3 = (bf16*)alloc((size_t)BT*544*2);
    bf16* Ain4 = (bf16*)alloc((size_t)BT*544*2);
    bf16* Ain5 = (bf16*)alloc((size_t)BT*512*2);

    char* gruState = alloc(BATCH*HDIM*4 + 2*BATCH*HDIM*2 + 256);
    float* hG   = (float*)gruState;
    bf16*  hGb0 = (bf16*)(gruState + BATCH*HDIM*4);
    bf16*  hGb1 = (bf16*)(gruState + BATCH*HDIM*4 + BATCH*HDIM*2);
    unsigned* cnt = (unsigned*)(gruState + BATCH*HDIM*4 + 2*BATCH*HDIM*2);
    size_t gruStateBytes = BATCH*HDIM*4 + 2*BATCH*HDIM*2 + 256;

    const int TB = 256;

    // ---- mel resnet (fp32, tiny) ----
    k_conv_in<<<blocks_for(BATCH*CCH*LCONV, TB), TB, 0, stream>>>(mels, conv_in_w, bn0_g, bn0_b, bufA);
    float* cur = bufA; float* nxt = bufB;
    for (int i = 0; i < 10; i++) {
        k_mm128<<<blocks_for(BATCH*CCH*LCONV, TB), TB, 0, stream>>>(
            cur, res_c1 + (size_t)i*CCH*CCH, rbn1g + i*CCH, rbn1b + i*CCH, nullptr, bufT, 1);
        k_mm128<<<blocks_for(BATCH*CCH*LCONV, TB), TB, 0, stream>>>(
            bufT, res_c2 + (size_t)i*CCH*CCH, rbn2g + i*CCH, rbn2b + i*CCH, cur, nxt, 0);
        float* t2 = cur; cur = nxt; nxt = t2;
    }
    k_mm128<<<blocks_for(BATCH*CCH*LCONV, TB), TB, 0, stream>>>(
        cur, conv_out_w, nullptr, conv_out_b, nullptr, aux_s, 0);

    // ---- upsample (fp32, tiny) ----
    k_upsample<<<blocks_for((size_t)BATCH*FEAT*216, TB), TB, 0, stream>>>(mels, u1, up_w0, FRAMES, 216, 4, 9, 0, 0);
    k_upsample<<<blocks_for((size_t)BATCH*FEAT*1728, TB), TB, 0, stream>>>(u1, u2, up_w1, 216, 1728, 8, 17, 0, 0);
    k_upsample<<<blocks_for((size_t)BATCH*FEAT*TLEN, TB), TB, 0, stream>>>(u2, m_up, up_w2, 1728, TLEN, 8, 17, 512, 1);

    // ---- weight conversion / packing ----
    k_cvt_w<<<blocks_for((size_t)512*128, TB), TB, 0, stream>>>(I_w, Iw_b, 512, 113, 128);
    k_cvt_w<<<blocks_for((size_t)1536*512, TB), TB, 0, stream>>>(r1_wih, r1wih_b, 1536, 512, 512);
    k_cvt_w<<<blocks_for((size_t)1536*544, TB), TB, 0, stream>>>(r2_wih, r2wih_b, 1536, 544, 544);
    k_cvt_w<<<blocks_for((size_t)512*544, TB), TB, 0, stream>>>(fc1_w, fc1w_b, 512, 544, 544);
    k_cvt_w<<<blocks_for((size_t)512*544, TB), TB, 0, stream>>>(fc2_w, fc2w_b, 512, 544, 544);
    k_cvt_w<<<blocks_for((size_t)1024*512, TB), TB, 0, stream>>>(fc3_w, fc3w_b, 1024, 512, 512);
    k_pack_whh<<<blocks_for((size_t)GRU_NWG*WHH_WG_ELEMS, TB), TB, 0, stream>>>(r1_whh, whh1P);
    k_pack_whh<<<blocks_for((size_t)GRU_NWG*WHH_WG_ELEMS, TB), TB, 0, stream>>>(r2_whh, whh2P);

    // ---- pack activations ----
    k_ain1<<<blocks_for((size_t)BT*128, TB), TB, 0, stream>>>(x, m_up, aux_s, Ain1);
    k_packaux<<<blocks_for((size_t)BT*AUXD, TB), TB, 0, stream>>>(aux_s, Ain2, 544, 1);
    k_packaux<<<blocks_for((size_t)BT*AUXD, TB), TB, 0, stream>>>(aux_s, Ain3, 544, 2);
    k_packaux<<<blocks_for((size_t)BT*AUXD, TB), TB, 0, stream>>>(aux_s, Ain4, 544, 3);

    // ---- GEMM1: h0 = Ain1 @ I_w^T + I_b ----
    k_gemm<<<(BT/32)*(512/64)/8, 256, 0, stream>>>(Ain1, Iw_b, I_b, nullptr, h0, BT, 512, 128, 512, 0);
    // ---- GEMM2: xp1 = h0 @ r1_wih^T + r1_bih ----
    k_gemm<<<(BT/32)*(1536/64)/8, 256, 0, stream>>>(h0, r1wih_b, r1_bih, xp, nullptr, BT, 1536, 512, 1536, 0);

    // ---- GRU1 (LDS-resident weights, writes h+res into Ain2 cols 0..511) ----
    hipMemsetAsync(gruState, 0, gruStateBytes, stream);
    k_gru<<<GRU_NWG, GRU_THREADS, GRU_LDS_BYTES, stream>>>(xp, whh1P, r1_bhh,
                                                           h0, 512, Ain2, 544,
                                                           hG, hGb0, hGb1, cnt, TLEN);

    // ---- GEMM3: xp2 = Ain2 @ r2_wih^T + r2_bih ----
    k_gemm<<<(BT/32)*(1536/64)/8, 256, 0, stream>>>(Ain2, r2wih_b, r2_bih, xp, nullptr, BT, 1536, 544, 1536, 0);

    // ---- GRU2 (writes h+res into Ain3 cols 0..511) ----
    hipMemsetAsync(gruState, 0, gruStateBytes, stream);
    k_gru<<<GRU_NWG, GRU_THREADS, GRU_LDS_BYTES, stream>>>(xp, whh2P, r2_bhh,
                                                           Ain2, 544, Ain3, 544,
                                                           hG, hGb0, hGb1, cnt, TLEN);

    // ---- fc1 / fc2 / fc3 ----
    k_gemm<<<(BT/32)*(512/64)/8, 256, 0, stream>>>(Ain3, fc1w_b, fc1_b, nullptr, Ain4, BT, 512, 544, 544, 1);
    k_gemm<<<(BT/32)*(512/64)/8, 256, 0, stream>>>(Ain4, fc2w_b, fc2_b, nullptr, Ain5, BT, 512, 544, 512, 1);
    k_gemm<<<(BT/32)*(1024/64)/8, 256, 0, stream>>>(Ain5, fc3w_b, fc3_b, (float*)d_out, nullptr, BT, 1024, 512, 1024, 0);
}
